// NodeUpdateV2_33827162423513
// MI455X (gfx1250) — compile-verified, matrix operand order checked by reading, not run on hardware
//
#include <hip/hip_runtime.h>
#include <hip/hip_bf16.h>

// ---------------------------------------------------------------------------
// CDNA5 (gfx1250): bf16 WMMA (16x16x32, f32 accum) for all GEMM stages,
// TDM (tensor_load_to_lds) for LDS weight staging, global f32 atomics for
// scatter-mean.
// ---------------------------------------------------------------------------

typedef __bf16        v16bf __attribute__((ext_vector_type(16)));
typedef float         v8f   __attribute__((ext_vector_type(8)));
typedef unsigned int  v4u   __attribute__((ext_vector_type(4)));
typedef unsigned int  v8u   __attribute__((ext_vector_type(8)));

#define NODE_DIM 128
#define EDGE_DIM 64
#define MSG_DIM  128
#define OUT_DIM  128
#define N_REL    8
#define K_MSG    (NODE_DIM + EDGE_DIM)          // 192
#define K_UNIT   (NODE_DIM + 2 * MSG_DIM)       // 384
#define K_ATTR   (NODE_DIM + MSG_DIM)           // 256

// ---- WMMA helpers ---------------------------------------------------------

__device__ __forceinline__ v8f wmma_bf16(v16bf a, v16bf b, v8f c) {
  return __builtin_amdgcn_wmma_f32_16x16x32_bf16(
      /*neg_a=*/false, a, /*neg_b=*/false, b,
      /*c_mod=*/(short)0, c, /*reuse_a=*/false, /*reuse_b=*/false);
}

// A-matrix 16x32 bf16 fragment gathered from an f32 row pointer.
// ISA 7.12.2: lane L holds row M = L%16;
//   lanes 0-15 : elems 0-7 -> K=0..7,  elems 8-15 -> K=16..23
//   lanes 16-31: elems 0-7 -> K=8..15, elems 8-15 -> K=24..31
__device__ __forceinline__ v16bf a_frag_f32(const float* rowp, int lane, float scale) {
  const int kh = ((lane >> 4) & 1) * 8;
  const float* p0 = rowp + kh;
  const float* p1 = p0 + 16;
  v16bf a;
#pragma unroll
  for (int i = 0; i < 8; ++i) a[i] = (__bf16)(p0[i] * scale);
#pragma unroll
  for (int i = 0; i < 8; ++i) a[i + 8] = (__bf16)(p1[i] * scale);
  return a;
}

// B-matrix 32x16 bf16 fragment from pre-transposed W^T stored [N][K] bf16.
// Lane L holds col N = L%16; K striping symmetric to the A fragment.
__device__ __forceinline__ v16bf b_frag_bf16(const unsigned short* Wt, int KK,
                                             int k0, int lane) {
  const int n  = lane & 15;
  const int kh = ((lane >> 4) & 1) * 8;
  const unsigned short* p = Wt + (size_t)n * KK + k0 + kh;
  v16bf b;
#pragma unroll
  for (int i = 0; i < 8; ++i) b[i] = __builtin_bit_cast(__bf16, p[i]);
#pragma unroll
  for (int i = 0; i < 8; ++i) b[i + 8] = __builtin_bit_cast(__bf16, p[i + 16]);
  return b;
}

// ---- Tensor Data Mover: 1-D tile load global -> LDS -----------------------
// Builds a D# (group0: 4 SGPRs, group1: 8 SGPRs) for a flat run of `bytes`
// bytes (8-byte elements) and issues tensor_load_to_lds via inline asm
// (portable across both gfx1250 toolchains).  Caller: one wave issues,
// then s_wait_tensorcnt 0 + block barrier.
__device__ __forceinline__ void tdm_load_to_lds_1d(void* lds_ptr,
                                                   const void* gptr,
                                                   unsigned bytes) {
  const unsigned lds_off = (unsigned)(unsigned long long)lds_ptr;   // addr[31:0] = LDS offset
  const unsigned long long ga = (unsigned long long)gptr;
  const unsigned n8 = bytes >> 3;            // 8-byte elements
  v4u g0;
  g0[0] = 1u;                                         // count=1, user descriptor
  g0[1] = lds_off;                                    // lds_addr
  g0[2] = (unsigned)ga;                               // global_addr[31:0]
  g0[3] = (unsigned)((ga >> 32) & 0x01FFFFFFu) | 0x80000000u; // addr[56:32] | type=2
  v8u g1;
  g1[0] = 3u << 16;                                   // data_size = 8 bytes
  g1[1] = (n8 & 0xFFFFu) << 16;                       // tensor_dim0[15:0]
  g1[2] = (n8 >> 16) | (1u << 16);                    // tensor_dim0[31:16] | tensor_dim1=1
  g1[3] = (n8 & 0xFFFFu) << 16;                       // tile_dim0 = n8
  g1[4] = 0u;                                         // tile_dim1=0 (1-D), tile_dim2=0
  g1[5] = n8;                                         // tensor_dim0_stride[31:0]
  g1[6] = 0u;
  g1[7] = 0u;
  asm volatile("tensor_load_to_lds %0, %1" :: "s"(g0), "s"(g1) : "memory");
}

// ---- utility kernels ------------------------------------------------------

__global__ void k_zero(float* p, long long n) {
  long long i = (long long)blockIdx.x * blockDim.x + threadIdx.x;
  if (i < n) p[i] = 0.0f;
}

// Convert + transpose all weights to bf16 W^T ([out][in]) in workspace.
__global__ void k_prep(const float* W_msg, const float* W_rel,
                       const float* W_unit, const float* W_attr,
                       unsigned short* Wt_msg, unsigned short* Wt_rel,
                       unsigned short* Wt_unit, unsigned short* Wt_attr) {
  const int T_MSG  = MSG_DIM * K_MSG;
  const int T_REL  = N_REL * MSG_DIM * NODE_DIM;
  const int T_UNIT = OUT_DIM * K_UNIT;
  const int T_ATTR = OUT_DIM * K_ATTR;
  int idx = blockIdx.x * blockDim.x + threadIdx.x;
  float f;
  unsigned short* dst;
  if (idx < T_MSG) {
    int n = idx / K_MSG, k = idx % K_MSG;
    f = W_msg[k * MSG_DIM + n];
    dst = Wt_msg + idx;
  } else if (idx < T_MSG + T_REL) {
    int j = idx - T_MSG;
    int r = j / (MSG_DIM * NODE_DIM);
    int t = j % (MSG_DIM * NODE_DIM);
    int n = t / NODE_DIM, k = t % NODE_DIM;
    f = W_rel[r * NODE_DIM * MSG_DIM + k * MSG_DIM + n];
    dst = Wt_rel + j;
  } else if (idx < T_MSG + T_REL + T_UNIT) {
    int j = idx - T_MSG - T_REL;
    int n = j / K_UNIT, k = j % K_UNIT;
    f = W_unit[k * OUT_DIM + n];
    dst = Wt_unit + j;
  } else if (idx < T_MSG + T_REL + T_UNIT + T_ATTR) {
    int j = idx - T_MSG - T_REL - T_UNIT;
    int n = j / K_ATTR, k = j % K_ATTR;
    f = W_attr[k * OUT_DIM + n];
    dst = Wt_attr + j;
  } else {
    return;
  }
  *dst = __builtin_bit_cast(unsigned short, (__bf16)f);
}

// ---- stage 1: attribute messages -----------------------------------------
__global__ void __launch_bounds__(128)
k_attr(const float* __restrict__ node_emb, const float* __restrict__ edge_emb,
       const int* __restrict__ eidx, const float* __restrict__ b_msg,
       const unsigned short* __restrict__ Wt_msg,
       float* __restrict__ attr_sum, float* __restrict__ attr_cnt, int E) {
  __shared__ __align__(16) unsigned short sW[MSG_DIM * K_MSG];   // W_msg^T, 48 KB
  __shared__ int sDst[4][16];

  // TDM stage of W_msg^T into LDS: one DMA from wave 0
  if (threadIdx.x < 32) {
    tdm_load_to_lds_1d(sW, Wt_msg, MSG_DIM * K_MSG * 2);
    __builtin_amdgcn_s_wait_tensorcnt(0);
  }
  __syncthreads();

  const int wave = threadIdx.x >> 5;
  const int lane = threadIdx.x & 31;
  const int tile = blockIdx.x * 4 + wave;
  const long long tb = (long long)tile * 16;
  if (tb >= E) return;
  const int valid = min(16, (int)(E - tb));
  const bool full = (valid == 16);

  const int* srcIdx = eidx;
  const int* dstIdx = eidx + E;
  if (lane < 16) sDst[wave][lane] = dstIdx[(int)tb + min(lane, valid - 1)];

  const int m  = lane & 15;
  const int em = (int)tb + min(m, valid - 1);
  const int s  = srcIdx[em];
  const float* nodep = node_emb + (long long)s * NODE_DIM;
  const float* edgep = edge_emb + (long long)em * EDGE_DIM;
  __builtin_prefetch(edge_emb + ((long long)em + 16) * EDGE_DIM, 0, 1);

  // A fragments for this edge tile (K = 192 -> 6 fragments), kept in VGPRs
  v16bf afr[K_MSG / 32];
#pragma unroll
  for (int ks = 0; ks < K_MSG / 32; ++ks) {
    const int kk = ks * 32;
    const float* ap = (kk < NODE_DIM) ? (nodep + kk) : (edgep + (kk - NODE_DIM));
    afr[ks] = a_frag_f32(ap, lane, 1.0f);
  }

#pragma unroll 1
  for (int n0 = 0; n0 < MSG_DIM; n0 += 16) {
    // preload all B fragments of this n-tile, then run the WMMA chain
    v16bf bfr[K_MSG / 32];
#pragma unroll
    for (int ks = 0; ks < K_MSG / 32; ++ks)
      bfr[ks] = b_frag_bf16(sW + (size_t)n0 * K_MSG, K_MSG, ks * 32, lane);
    v8f acc = {0.f, 0.f, 0.f, 0.f, 0.f, 0.f, 0.f, 0.f};
#pragma unroll
    for (int ks = 0; ks < K_MSG / 32; ++ks) acc = wmma_bf16(afr[ks], bfr[ks], acc);

    const float bn = b_msg[n0 + (lane & 15)];
    const int   nc = n0 + (lane & 15);
    if (full) {
#pragma unroll
      for (int j = 0; j < 8; ++j) {
        const int row = j + ((lane >> 4) << 3);
        const float v = fmaxf(acc[j] + bn, 0.0f);
        atomicAdd(&attr_sum[(long long)sDst[wave][row] * MSG_DIM + nc], v);
      }
    } else {
#pragma unroll
      for (int j = 0; j < 8; ++j) {
        const int row = j + ((lane >> 4) << 3);
        if (row < valid) {
          const float v = fmaxf(acc[j] + bn, 0.0f);
          atomicAdd(&attr_sum[(long long)sDst[wave][row] * MSG_DIM + nc], v);
        }
      }
    }
  }
  if (lane < valid) atomicAdd(&attr_cnt[dstIdx[(int)tb + lane]], 1.0f);
}

// ---- stage 2a: per-(node, relation) edge counts ---------------------------
__global__ void k_relcnt(const int* __restrict__ eidx, const int* __restrict__ etype,
                         float* __restrict__ rel_cnt, int E) {
  int e = blockIdx.x * blockDim.x + threadIdx.x;
  if (e < E) atomicAdd(&rel_cnt[(long long)eidx[E + e] * N_REL + etype[e]], 1.0f);
}

// ---- stage 2b: relational messages ---------------------------------------
__global__ void __launch_bounds__(128)
k_rel(const float* __restrict__ node_emb, const int* __restrict__ eidx,
      const int* __restrict__ etype, const float* __restrict__ b_rel,
      const unsigned short* __restrict__ Wt_rel, const float* __restrict__ rel_cnt,
      float* __restrict__ rel_accum, int E) {
  __shared__ __align__(16) unsigned short sW[MSG_DIM * NODE_DIM]; // one W_rel[r]^T, 32 KB
  __shared__ int      sDst[4][16];
  __shared__ float    sWgt[4][16];
  __shared__ unsigned sPresent;        // block-wide union of types
  __shared__ unsigned sWaveMask[4];    // per-wave union of types

  const int wave = threadIdx.x >> 5;
  const int lane = threadIdx.x & 31;
  const int tile = blockIdx.x * 4 + wave;
  const long long tb = (long long)tile * 16;
  const bool tileValid = tb < E;
  const int valid = tileValid ? min(16, (int)(E - tb)) : 0;

  if (threadIdx.x == 0) sPresent = 0u;
  if (lane == 0) sWaveMask[wave] = 0u;
  __syncthreads();
  if (tileValid && lane < valid) {
    const unsigned bit = 1u << etype[(int)tb + lane];
    atomicOr(&sPresent, bit);
    atomicOr(&sWaveMask[wave], bit);
  }
  __syncthreads();
  const unsigned present  = sPresent;        // block-uniform
  const unsigned waveMask = sWaveMask[wave]; // wave-uniform

  const int* srcIdx = eidx;
  const int* dstIdx = eidx + E;
  const int m  = lane & 15;
  const int em = tileValid ? (int)tb + min(m, valid - 1) : 0;
  const float* nodep = node_emb + (long long)srcIdx[em] * NODE_DIM;

  // A fragments (K = 128 -> 4 fragments), reused for every relation type
  v16bf afr[NODE_DIM / 32];
#pragma unroll
  for (int ks = 0; ks < NODE_DIM / 32; ++ks)
    afr[ks] = a_frag_f32(nodep + ks * 32, lane, 1.0f);

  for (int r = 0; r < N_REL; ++r) {
    if (!((present >> r) & 1u)) continue;    // block-uniform
    __syncthreads();                         // prior readers of sW done
    if (threadIdx.x < 32) {                  // TDM stage W_rel[r]^T
      tdm_load_to_lds_1d(sW, Wt_rel + (size_t)r * MSG_DIM * NODE_DIM,
                         MSG_DIM * NODE_DIM * 2);
      __builtin_amdgcn_s_wait_tensorcnt(0);
    }
    if (tileValid && lane < 16) {
      const int e = (int)tb + min(lane, valid - 1);
      float w = 0.0f;
      const int d = dstIdx[e];
      if (lane < valid && etype[e] == r)
        w = 1.0f / fmaxf(rel_cnt[(long long)d * N_REL + r], 1.0f);
      sWgt[wave][lane] = w;
      sDst[wave][lane] = d;
    }
    __syncthreads();
    if (tileValid && ((waveMask >> r) & 1u)) {
#pragma unroll 1
      for (int n0 = 0; n0 < MSG_DIM; n0 += 16) {
        v16bf bfr[NODE_DIM / 32];
#pragma unroll
        for (int ks = 0; ks < NODE_DIM / 32; ++ks)
          bfr[ks] = b_frag_bf16(sW + (size_t)n0 * NODE_DIM, NODE_DIM, ks * 32, lane);
        v8f acc = {0.f, 0.f, 0.f, 0.f, 0.f, 0.f, 0.f, 0.f};
#pragma unroll
        for (int ks = 0; ks < NODE_DIM / 32; ++ks)
          acc = wmma_bf16(afr[ks], bfr[ks], acc);

        const float bn = b_rel[r * MSG_DIM + n0 + (lane & 15)];
        const int   nc = n0 + (lane & 15);
#pragma unroll
        for (int j = 0; j < 8; ++j) {
          const int row = j + ((lane >> 4) << 3);
          const float w = sWgt[wave][row];
          if (w != 0.0f) {
            const float v = fmaxf(acc[j] + bn, 0.0f) * w;
            atomicAdd(&rel_accum[(long long)sDst[wave][row] * MSG_DIM + nc], v);
          }
        }
      }
    }
  }
}

// ---- stage 3: node update -------------------------------------------------
__global__ void __launch_bounds__(128)
k_node(const float* __restrict__ node_emb, const float* __restrict__ attr_sum,
       const float* __restrict__ attr_cnt, const float* __restrict__ rel_accum,
       const unsigned short* __restrict__ Wt_unit,
       const unsigned short* __restrict__ Wt_attr,
       const float* __restrict__ b_unit, const float* __restrict__ b_attr,
       const int* __restrict__ is_unit, float* __restrict__ out, int N) {
  const int wave = threadIdx.x >> 5;
  const int lane = threadIdx.x & 31;
  const int tile = blockIdx.x * 4 + wave;
  const long long tb = (long long)tile * 16;
  if (tb >= N) return;
  const int valid = min(16, (int)(N - tb));

  const int m = lane & 15;
  const int v = (int)tb + min(m, valid - 1);
  const float* np = node_emb  + (long long)v * NODE_DIM;
  const float* ap = attr_sum  + (long long)v * MSG_DIM;
  const float* rp = rel_accum + (long long)v * MSG_DIM;
  const float ascale = 1.0f / fmaxf(attr_cnt[v], 1.0f);

#pragma unroll 1
  for (int n0 = 0; n0 < OUT_DIM; n0 += 16) {
    v8f accU = {0.f, 0.f, 0.f, 0.f, 0.f, 0.f, 0.f, 0.f};
    v8f accA = {0.f, 0.f, 0.f, 0.f, 0.f, 0.f, 0.f, 0.f};
#pragma unroll
    for (int ks = 0; ks < K_UNIT / 32; ++ks) {
      const int kk = ks * 32;
      const float* p;
      float sc = 1.0f;
      if (kk < NODE_DIM) {
        p = np + kk;
      } else if (kk < NODE_DIM + MSG_DIM) {
        p = ap + (kk - NODE_DIM);
        sc = ascale;
      } else {
        p = rp + (kk - NODE_DIM - MSG_DIM);
      }
      v16bf a  = a_frag_f32(p, lane, sc);
      v16bf bu = b_frag_bf16(Wt_unit + (size_t)n0 * K_UNIT, K_UNIT, kk, lane);
      accU = wmma_bf16(a, bu, accU);
      if (kk < K_ATTR) {
        v16bf ba = b_frag_bf16(Wt_attr + (size_t)n0 * K_ATTR, K_ATTR, kk, lane);
        accA = wmma_bf16(a, ba, accA);
      }
    }
    const int n = n0 + (lane & 15);
    const float bu = b_unit[n];
    const float ba = b_attr[n];
#pragma unroll
    for (int j = 0; j < 8; ++j) {
      const int row = j + ((lane >> 4) << 3);
      if (row < valid) {
        const int node = (int)tb + row;
        const float hu = fmaxf(accU[j] + bu, 0.0f);
        const float ha = fmaxf(accA[j] + ba, 0.0f);
        out[(long long)node * OUT_DIM + n] = is_unit[node] ? hu : ha;
      }
    }
  }
}

// ---------------------------------------------------------------------------

extern "C" void kernel_launch(void* const* d_in, const int* in_sizes, int n_in,
                              void* d_out, int out_size, void* d_ws, size_t ws_size,
                              hipStream_t stream) {
  const float* node_emb = (const float*)d_in[0];
  const float* edge_emb = (const float*)d_in[1];
  const float* W_msg    = (const float*)d_in[2];
  const float* b_msg    = (const float*)d_in[3];
  const float* W_rel    = (const float*)d_in[4];
  const float* b_rel    = (const float*)d_in[5];
  const float* W_unit   = (const float*)d_in[6];
  const float* b_unit   = (const float*)d_in[7];
  const float* W_attr   = (const float*)d_in[8];
  const float* b_attr   = (const float*)d_in[9];
  const int* data_ei    = (const int*)d_in[10];
  const int* rel_ei     = (const int*)d_in[11];
  const int* rel_type   = (const int*)d_in[12];
  const int* is_unit    = (const int*)d_in[13];

  const int N  = in_sizes[0] / NODE_DIM;
  const int EA = in_sizes[1] / EDGE_DIM;
  const int ER = in_sizes[12];

  // workspace layout
  char* ws = (char*)d_ws;
  size_t o = 0;
  float* attr_sum  = (float*)(ws + o); o += (size_t)N * MSG_DIM * 4;
  float* rel_accum = (float*)(ws + o); o += (size_t)N * MSG_DIM * 4;
  float* attr_cnt  = (float*)(ws + o); o += (size_t)N * 4;
  float* rel_cnt   = (float*)(ws + o); o += (size_t)N * N_REL * 4;
  o = (o + 255) & ~(size_t)255;
  unsigned short* Wt_msg  = (unsigned short*)(ws + o); o += (size_t)MSG_DIM * K_MSG * 2;
  unsigned short* Wt_rel  = (unsigned short*)(ws + o); o += (size_t)N_REL * MSG_DIM * NODE_DIM * 2;
  unsigned short* Wt_unit = (unsigned short*)(ws + o); o += (size_t)OUT_DIM * K_UNIT * 2;
  unsigned short* Wt_attr = (unsigned short*)(ws + o); o += (size_t)OUT_DIM * K_ATTR * 2;

  {
    long long zn = (long long)N * (2 * MSG_DIM + 1 + N_REL);
    k_zero<<<(int)((zn + 255) / 256), 256, 0, stream>>>(attr_sum, zn);
  }
  {
    const int T = MSG_DIM * K_MSG + N_REL * MSG_DIM * NODE_DIM +
                  OUT_DIM * K_UNIT + OUT_DIM * K_ATTR;
    k_prep<<<(T + 255) / 256, 256, 0, stream>>>(W_msg, W_rel, W_unit, W_attr,
                                                Wt_msg, Wt_rel, Wt_unit, Wt_attr);
  }
  {
    int tiles = (EA + 15) / 16;
    k_attr<<<(tiles + 3) / 4, 128, 0, stream>>>(node_emb, edge_emb, data_ei, b_msg,
                                                Wt_msg, attr_sum, attr_cnt, EA);
  }
  k_relcnt<<<(ER + 255) / 256, 256, 0, stream>>>(rel_ei, rel_type, rel_cnt, ER);
  {
    int tiles = (ER + 15) / 16;
    k_rel<<<(tiles + 3) / 4, 128, 0, stream>>>(node_emb, rel_ei, rel_type, b_rel,
                                               Wt_rel, rel_cnt, rel_accum, ER);
  }
  {
    int tiles = (N + 15) / 16;
    k_node<<<(tiles + 3) / 4, 128, 0, stream>>>(node_emb, attr_sum, attr_cnt,
                                                rel_accum, Wt_unit, Wt_attr,
                                                b_unit, b_attr, is_unit,
                                                (float*)d_out, N);
  }
}
